// GCN_Block_52810917872369
// MI455X (gfx1250) — matrix operand order repaired and checked by reading, not compiled, not run on hardware
//
#include <hip/hip_runtime.h>
#include <hip/hip_bf16.h>

#define BN   32
#define CIN  64
#define COUT 128
#define PPART 3
#define TT   300
#define VV   25
#define RR   (TT*VV)          // 7500
#define MALL (BN*RR)          // 240000
#define KCAT 576              // 384 (temporal im2col) + 192 (residual im2col)
#define EPSB 1e-5f

typedef __attribute__((ext_vector_type(16))) __bf16 v16bf;
typedef __attribute__((ext_vector_type(8)))  float  f32x8;
typedef __attribute__((ext_vector_type(4)))  unsigned int u32x4;
typedef __attribute__((ext_vector_type(8)))  int i32x8;
typedef __attribute__((ext_vector_type(4)))  int i32x4;

union Frag16 { v16bf v; uint4 q[2]; __bf16 e[16]; };
union Pack8  { __bf16 e[8]; uint4 q; };
union F8U    { f32x8 v; uint4 q[2]; };

__device__ __forceinline__ uint4 ld16(const __bf16* p) {
  return *reinterpret_cast<const uint4*>(p);
}
__device__ __forceinline__ f32x8 wmma_bf16(const Frag16& a, const Frag16& b, f32x8 c) {
  return __builtin_amdgcn_wmma_f32_16x16x32_bf16(false, a.v, false, b.v,
                                                 (short)0, c, false, false);
}

// ---------------------------------------------------------------------------
// Kernel 1: fold BN into weights, build bf16 B-matrices in [N][K] layout.
// ---------------------------------------------------------------------------
__global__ void prep_kernel(const float* __restrict__ Aadj, const float* __restrict__ EI,
                            const float* __restrict__ w1, const float* __restrict__ b1,
                            const float* __restrict__ g1, const float* __restrict__ be1,
                            const float* __restrict__ m1, const float* __restrict__ v1,
                            const float* __restrict__ g2, const float* __restrict__ be2,
                            const float* __restrict__ m2, const float* __restrict__ v2,
                            const float* __restrict__ wt, const float* __restrict__ bt,
                            const float* __restrict__ g3, const float* __restrict__ be3,
                            const float* __restrict__ m3, const float* __restrict__ v3,
                            const float* __restrict__ wr, const float* __restrict__ br,
                            const float* __restrict__ gr, const float* __restrict__ ber,
                            const float* __restrict__ mr, const float* __restrict__ vr,
                            __bf16* __restrict__ WcfT,  // [128][576]
                            __bf16* __restrict__ W1f,   // [384][64]
                            __bf16* __restrict__ EWfT,  // [32][96]
                            float* __restrict__ b1f,    // [384]
                            float* __restrict__ bcf,    // [128]
                            float* __restrict__ sc2,    // [128]
                            float* __restrict__ sh2)    // [128]
{
  int i = blockIdx.x * blockDim.x + threadIdx.x;
  if (i < 128*KCAT) {                       // fused temporal+residual B matrix
    int c = i / KCAT, k = i % KCAT;
    float val;
    if (k < 384) {
      int dt = k >> 7, cin = k & 127;
      float s3 = g3[c] * rsqrtf(v3[c] + EPSB);
      val = wt[(c*COUT + cin)*3 + dt] * s3;
    } else {
      int k2 = k - 384, dt = k2 >> 6, cin = k2 & 63;
      float sR = gr[c] * rsqrtf(vr[c] + EPSB);
      val = wr[(c*CIN + cin)*3 + dt] * sR;
    }
    WcfT[i] = (__bf16)val;
    return;
  }
  i -= 128*KCAT;
  if (i < 384*64) {                         // 1x1-conv B matrix (BN1 folded)
    int c = i / 64;
    float s1 = g1[c] * rsqrtf(v1[c] + EPSB);
    W1f[i] = (__bf16)(w1[i] * s1);
    return;
  }
  i -= 384*64;
  if (i < 32*96) {                          // edge weights, padded [32][96]
    int w = i / 96, pj = i % 96;
    float val = 0.f;
    if (w < VV && pj < PPART*VV) {
      int p = pj / VV, j = pj % VV;
      int a = p*VV*VV + j*VV + w;
      val = Aadj[a] * EI[a];
    }
    EWfT[i] = (__bf16)val;
    return;
  }
  i -= 32*96;
  if (i < 384) {                            // folded bias of conv1+BN1
    float s1 = g1[i] * rsqrtf(v1[i] + EPSB);
    b1f[i] = (b1[i] - m1[i]) * s1 + be1[i];
    return;
  }
  i -= 384;
  if (i < 128) {                            // combined temporal+residual bias, BN2 affine
    float s3 = g3[i] * rsqrtf(v3[i] + EPSB);
    float sR = gr[i] * rsqrtf(vr[i] + EPSB);
    bcf[i] = (bt[i] - m3[i]) * s3 + be3[i] + (br[i] - mr[i]) * sR + ber[i];
    float s2 = g2[i] * rsqrtf(v2[i] + EPSB);
    sc2[i] = s2;
    sh2[i] = be2[i] - m2[i] * s2;
  }
}

// ---------------------------------------------------------------------------
// Kernel 2: NCHW f32 x -> Xt bf16 [pos = b*7500 + t*25 + v][64] (A-friendly).
// ---------------------------------------------------------------------------
__global__ void transpose_kernel(const float* __restrict__ x, __bf16* __restrict__ Xt) {
  int pos = blockIdx.x * blockDim.x + threadIdx.x;
  if (pos >= MALL) return;
  int b = pos / RR, r = pos % RR;
  const float* xp = x + (long)b * CIN * RR + r;
  union { __bf16 e[64]; uint4 q[8]; } buf;
#pragma unroll
  for (int k = 0; k < CIN; ++k) buf.e[k] = (__bf16)xp[k * RR];
  uint4* dst = reinterpret_cast<uint4*>(Xt + (long)pos * CIN);
#pragma unroll
  for (int q = 0; q < 8; ++q) dst[q] = buf.q[q];
}

// ---------------------------------------------------------------------------
// Kernel 3 (fused): per (b,t):
//   TDM:    DMA the 25x64 bf16 Xt tile into LDS (tensor_load_to_lds)
//   stage1: Y1[j][pc] = relu(Xs[j][:] @ W1f^T + b1f)   (32x384, K=64) -> LDS
//   stage2: h2[c][w]  = relu(BN2( sum_pj Y1'[c][pj] * EW[pj][w] ))  (128x32, K=96)
// ---------------------------------------------------------------------------
__global__ void __launch_bounds__(256) gcn_spatial_kernel(
    const __bf16* __restrict__ Xt, const __bf16* __restrict__ W1f,
    const float* __restrict__ b1f, const __bf16* __restrict__ EWfT,
    const float* __restrict__ sc2, const float* __restrict__ sh2,
    __bf16* __restrict__ h2)
{
  __shared__ __bf16 Xs[VV][CIN];            // 25 x 64 bf16 = 3.2 KB (TDM dest)
  __shared__ __bf16 Y1s[VV][392];           // 25 x 384 (pad to 392)
  const int bt   = blockIdx.x;              // 0..9599 = b*300 + t
  const int wave = threadIdx.x >> 5;
  const int lane = threadIdx.x & 31;
  const int l16  = lane & 15, lhi = lane >> 4;

  // ---- TDM: async tile DMA Global -> LDS, issued by wave 0 only -----------
  if (wave == 0) {
    unsigned long long ga =
        (unsigned long long)(uintptr_t)(Xt + (size_t)bt * VV * CIN);
    unsigned int lds = (unsigned int)(uintptr_t)&Xs[0][0];
    // D# group0: count=1 | lds_addr | global_addr[56:0] | type=2 ("image")
    u32x4 g0 = { 1u, lds, (unsigned int)ga,
                 (unsigned int)(((ga >> 32) & 0x01FFFFFFu) | (2u << 30)) };
    // D# group1: data_size=1 (2B), tensor_dim0=64, tensor_dim1=25,
    //            tile_dim0=64, tile_dim1=25, tensor_dim0_stride=64
    i32x8 g1 = { 0x10000, (CIN << 16), (VV << 16), (CIN << 16),
                 VV, CIN, 0, 0 };
    i32x4 gz4 = { 0, 0, 0, 0 };             // 2D tensor: groups 2/3 unused
    i32x8 gz8 = { 0, 0, 0, 0, 0, 0, 0, 0 };
    __builtin_amdgcn_tensor_load_to_lds(g0, g1, gz4, gz4, gz8, 0);
    __builtin_amdgcn_s_wait_tensorcnt(0);
  }
  __syncthreads();

  // ---- stage 1: 1x1 conv, tiles: mt in {0,1}, nt in 0..23 -----------------
  {
    const int mt  = wave >> 2;
    const int ntb = wave & 3;
    f32x8 acc[6];
#pragma unroll
    for (int q = 0; q < 6; ++q) acc[q] = (f32x8){0,0,0,0,0,0,0,0};
    const int j = mt*16 + l16;              // row (vertex), valid if < 25
    const __bf16* Xrow = &Xs[0][0] + j * CIN;
#pragma unroll
    for (int ks = 0; ks < 2; ++ks) {
      Frag16 a;
      const int k0 = ks*32 + lhi*8;
      if (j < VV) { a.q[0] = ld16(Xrow + k0); a.q[1] = ld16(Xrow + k0 + 16); }
      else        { a.q[0] = (uint4){0,0,0,0}; a.q[1] = (uint4){0,0,0,0}; }
#pragma unroll
      for (int q = 0; q < 6; ++q) {
        const int nt = ntb + q*4;
        const int c  = nt*16 + l16;
        Frag16 bf;
        const __bf16* bp = W1f + c*CIN + ks*32 + lhi*8;
        bf.q[0] = ld16(bp); bf.q[1] = ld16(bp + 16);
        acc[q] = wmma_bf16(a, bf, acc[q]);
      }
    }
    // epilogue: bias + ReLU -> LDS (Y1s[j][c], bf16)
    const int jb = mt*16 + (lhi ? 8 : 0);
#pragma unroll
    for (int q = 0; q < 6; ++q) {
      const int nt = ntb + q*4;
      const int c  = nt*16 + l16;
      const float bias = b1f[c];
#pragma unroll
      for (int g = 0; g < 8; ++g) {
        const int jj = jb + g;
        if (jj < VV) Y1s[jj][c] = (__bf16)fmaxf(acc[q][g] + bias, 0.f);
      }
    }
  }
  __syncthreads();

  // ---- stage 2: graph aggregation, tiles: mt = wave (c), nt in {0,1} (w) --
  {
    const int mtc = wave;
    const int c   = mtc*16 + l16;           // A row (channel)
    f32x8 acc[2];
    acc[0] = (f32x8){0,0,0,0,0,0,0,0};
    acc[1] = (f32x8){0,0,0,0,0,0,0,0};
#pragma unroll
    for (int ks = 0; ks < 3; ++ks) {
      Frag16 a;
#pragma unroll
      for (int e = 0; e < 16; ++e) {
        const int kl = lhi*8 + e + (e >= 8 ? 8 : 0);
        const int pj = ks*32 + kl;
        __bf16 val = (__bf16)0.f;
        if (pj < PPART*VV) val = Y1s[pj % VV][(pj / VV)*COUT + c];
        a.e[e] = val;
      }
#pragma unroll
      for (int nt = 0; nt < 2; ++nt) {
        const int w = nt*16 + l16;
        Frag16 bf;
        const __bf16* bp = EWfT + w*96 + ks*32 + lhi*8;
        bf.q[0] = ld16(bp); bf.q[1] = ld16(bp + 16);
        acc[nt] = wmma_bf16(a, bf, acc[nt]);
      }
    }
    // epilogue: BN2 + ReLU -> h2[(bt*25+w)*128 + c], c contiguous per lane
    const int cb = mtc*16 + (lhi ? 8 : 0);
#pragma unroll
    for (int nt = 0; nt < 2; ++nt) {
      const int w = nt*16 + l16;
      if (w < VV) {
        Pack8 o;
#pragma unroll
        for (int g = 0; g < 8; ++g) {
          const int cc = cb + g;
          o.e[g] = (__bf16)fmaxf(acc[nt][g] * sc2[cc] + sh2[cc], 0.f);
        }
        *reinterpret_cast<uint4*>(h2 + ((long)bt*VV + w)*COUT + cb) = o.q;
      }
    }
  }
}

// ---------------------------------------------------------------------------
// Kernel 4 (fused): temporal conv + residual conv as one GEMM, K=576.
//   A[(b,r)][k]: k<384 -> h2[b][r+(dt-1)*25][cin], k>=384 -> Xt[b][r+(dt-1)*25][cin]
//   out[b][c][r] = acc + bcf[c]   (f32, NCHW)
// ---------------------------------------------------------------------------
__global__ void __launch_bounds__(256) gcn_temporal_kernel(
    const __bf16* __restrict__ h2, const __bf16* __restrict__ Xt,
    const __bf16* __restrict__ WcfT, const float* __restrict__ bcf,
    float* __restrict__ out)
{
  const int b    = blockIdx.y;
  const int r0   = blockIdx.x * 128;
  const int wave = threadIdx.x >> 5;
  const int lane = threadIdx.x & 31;
  const int l16  = lane & 15, lhi = lane >> 4;
  const int row  = r0 + wave*16 + l16;      // flat (t*25+v), may exceed 7499

  const __bf16* h2b = h2 + (long)b * RR * COUT;
  const __bf16* Xtb = Xt + (long)b * RR * CIN;

  f32x8 acc[8];
#pragma unroll
  for (int q = 0; q < 8; ++q) acc[q] = (f32x8){0,0,0,0,0,0,0,0};

  for (int ks = 0; ks < KCAT/32; ++ks) {
    const int k0 = ks*32;
    Frag16 a;
#pragma unroll
    for (int h = 0; h < 2; ++h) {
      const int kh = k0 + lhi*8 + h*16;     // 8 contiguous k -> one b128 load
      uint4 qv = {0,0,0,0};
      if (kh < 384) {
        const int dt = kh >> 7, cin = kh & 127;
        const int rr = row + (dt - 1)*VV;
        if (row < RR && rr >= 0 && rr < RR) {
          qv = ld16(h2b + (long)rr*COUT + cin);
          __builtin_prefetch(h2b + (long)rr*COUT + cin + 32, 0, 1);
        }
      } else {
        const int k2 = kh - 384, dt = k2 >> 6, cin = k2 & 63;
        const int rr = row + (dt - 1)*VV;
        if (row < RR && rr >= 0 && rr < RR)
          qv = ld16(Xtb + (long)rr*CIN + cin);
      }
      a.q[h] = qv;
    }
#pragma unroll
    for (int nt = 0; nt < 8; ++nt) {
      const int c = nt*16 + l16;
      Frag16 bf;
      const __bf16* bp = WcfT + c*KCAT + k0 + lhi*8;
      bf.q[0] = ld16(bp); bf.q[1] = ld16(bp + 16);
      acc[nt] = wmma_bf16(a, bf, acc[nt]);
    }
  }

  // epilogue: rows are contiguous per lane -> 32B f32 stores into NCHW out
  const int rb = r0 + wave*16 + (lhi ? 8 : 0);
#pragma unroll
  for (int nt = 0; nt < 8; ++nt) {
    const int c = nt*16 + l16;
    const float bias = bcf[c];
    float* op = out + ((long)(b*COUT + c))*RR + rb;
    if (rb + 8 <= RR) {
      F8U o;
#pragma unroll
      for (int g = 0; g < 8; ++g) o.v[g] = acc[nt][g] + bias;
      reinterpret_cast<uint4*>(op)[0] = o.q[0];
      reinterpret_cast<uint4*>(op)[1] = o.q[1];
    } else {
#pragma unroll
      for (int g = 0; g < 8; ++g)
        if (rb + g < RR) op[g] = acc[nt][g] + bias;
    }
  }
}

// ---------------------------------------------------------------------------
extern "C" void kernel_launch(void* const* d_in, const int* in_sizes, int n_in,
                              void* d_out, int out_size, void* d_ws, size_t ws_size,
                              hipStream_t stream) {
  const float* x   = (const float*)d_in[0];
  const float* Aa  = (const float*)d_in[1];
  const float* EI  = (const float*)d_in[2];
  const float* w1  = (const float*)d_in[3];
  const float* b1  = (const float*)d_in[4];
  const float* g1  = (const float*)d_in[5];
  const float* be1 = (const float*)d_in[6];
  const float* m1  = (const float*)d_in[7];
  const float* v1  = (const float*)d_in[8];
  const float* g2  = (const float*)d_in[9];
  const float* be2 = (const float*)d_in[10];
  const float* m2  = (const float*)d_in[11];
  const float* v2  = (const float*)d_in[12];
  const float* wt  = (const float*)d_in[13];
  const float* bt  = (const float*)d_in[14];
  const float* g3  = (const float*)d_in[15];
  const float* be3 = (const float*)d_in[16];
  const float* m3  = (const float*)d_in[17];
  const float* v3  = (const float*)d_in[18];
  const float* wr  = (const float*)d_in[19];
  const float* br  = (const float*)d_in[20];
  const float* gr  = (const float*)d_in[21];
  const float* ber = (const float*)d_in[22];
  const float* mr  = (const float*)d_in[23];
  const float* vr  = (const float*)d_in[24];

  char* ws = (char*)d_ws;
  size_t off = 0;
  __bf16* Xt   = (__bf16*)(ws + off); off += (size_t)MALL * CIN * 2;      // 30.72 MB
  __bf16* h2   = (__bf16*)(ws + off); off += (size_t)MALL * COUT * 2;     // 61.44 MB
  __bf16* WcfT = (__bf16*)(ws + off); off += (size_t)COUT * KCAT * 2;
  __bf16* W1f  = (__bf16*)(ws + off); off += (size_t)384 * 64 * 2;
  __bf16* EWfT = (__bf16*)(ws + off); off += (size_t)32 * 96 * 2;
  float*  b1f  = (float*)(ws + off);  off += 384 * 4;
  float*  bcf  = (float*)(ws + off);  off += 128 * 4;
  float*  sc2  = (float*)(ws + off);  off += 128 * 4;
  float*  sh2  = (float*)(ws + off);  off += 128 * 4;

  const int prep_total = 128*KCAT + 384*64 + 32*96 + 384 + 128;
  prep_kernel<<<(prep_total + 255)/256, 256, 0, stream>>>(
      Aa, EI, w1, b1, g1, be1, m1, v1, g2, be2, m2, v2,
      wt, bt, g3, be3, m3, v3, wr, br, gr, ber, mr, vr,
      WcfT, W1f, EWfT, b1f, bcf, sc2, sh2);

  transpose_kernel<<<(MALL + 255)/256, 256, 0, stream>>>(x, Xt);

  gcn_spatial_kernel<<<BN*TT, 256, 0, stream>>>(Xt, W1f, b1f, EWfT, sc2, sh2, h2);

  gcn_temporal_kernel<<<dim3((RR + 127)/128, BN), 256, 0, stream>>>(
      h2, Xt, WcfT, bcf, (float*)d_out);
}
